// TransformerSA_25202868093374
// MI455X (gfx1250) — matrix operand order repaired and checked by reading, not compiled, not run on hardware
//
#include <hip/hip_runtime.h>

typedef _Float16 h16;
typedef __attribute__((ext_vector_type(16))) _Float16 v16h;
typedef __attribute__((ext_vector_type(8)))  _Float16 h16x8;
typedef __attribute__((ext_vector_type(8)))  float    v8f;
typedef __attribute__((ext_vector_type(4)))  int      i32x4;

#define EMB 1024
#define SEQ 2048
#define NH 16
#define HD 64

#if defined(__gfx1250__) && \
    __has_builtin(__builtin_amdgcn_global_load_async_to_lds_b128)
#define USE_ASYNC_LDS 1
#else
#define USE_ASYNC_LDS 0
#endif

// 16-byte tile copy: global -> LDS. Async (direct-to-LDS, ASYNCcnt) when the
// toolchain exposes the gfx1250 builtin, else VGPR bounce.
__device__ __forceinline__ void cp16(h16* l, const h16* g) {
#if USE_ASYNC_LDS
  h16* gg = const_cast<h16*>(g);
  __builtin_amdgcn_global_load_async_to_lds_b128(
      (__attribute__((address_space(1))) i32x4*)gg,
      (__attribute__((address_space(3))) i32x4*)l, 0, 0);
#else
  *(h16x8*)l = *(const h16x8*)g;
#endif
}

__device__ __forceinline__ void cp_wait() {
#if USE_ASYNC_LDS
#if __has_builtin(__builtin_amdgcn_s_wait_asynccnt)
  __builtin_amdgcn_s_wait_asynccnt(0);
#else
  asm volatile("s_wait_asynccnt 0" ::: "memory");
#endif
#endif
}

// ---------------------------------------------------------------------------
// WMMA fragment helpers (layouts per CDNA5 ISA 7.12.2, wave32)
// A 16x32 f16: lane l holds row m=l&15; elems 0..7 -> k=kb..kb+7, elems 8..15
// -> k=16+kb..16+kb+7 where kb = 8*(l>=16). Source row-major [m][k], ld elems.
__device__ __forceinline__ v16h load_a_frag(const h16* p0, int ld) {
  const int l  = threadIdx.x & 31;
  const int m  = l & 15;
  const int kb = (l >> 4) << 3;
  const h16* p = p0 + m * ld + kb;
  h16x8 lo = *(const h16x8*)(p);
  h16x8 hi = *(const h16x8*)(p + 16);
  v16h a;
#pragma unroll
  for (int i = 0; i < 8; ++i) { a[i] = lo[i]; a[i + 8] = hi[i]; }
  return a;
}

// B 32x16 f16: lane l holds col n=l&15; elems 0..7 -> k=kb..kb+7, elems 8..15
// -> k=16+kb..16+kb+7 where kb = 8*(l>=16).
// Source stored [n][k] (row n, contiguous k), ld elems.
__device__ __forceinline__ v16h load_b_frag(const h16* p0, int ld) {
  const int l  = threadIdx.x & 31;
  const int n  = l & 15;
  const int kb = (l >> 4) << 3;
  const h16* p = p0 + n * ld + kb;
  h16x8 lo = *(const h16x8*)(p);
  h16x8 hi = *(const h16x8*)(p + 16);
  v16h b;
#pragma unroll
  for (int i = 0; i < 8; ++i) { b[i] = lo[i]; b[i + 8] = hi[i]; }
  return b;
}

__device__ __forceinline__ v8f wmma_f16(v16h a, v16h b, v8f c) {
  return __builtin_amdgcn_wmma_f32_16x16x32_f16(false, a, false, b, (short)0, c,
                                                false, false);
}

// ---------------------------------------------------------------------------
__global__ void f32_to_f16_kernel(const float* __restrict__ src,
                                  h16* __restrict__ dst, int n) {
  int i = blockIdx.x * 256 + threadIdx.x;
  if (i < n) dst[i] = (h16)src[i];
}

// One block (256 threads) per row of 1024.
__global__ void rmsnorm_kernel(const float* __restrict__ x,
                               const float* __restrict__ w,
                               h16* __restrict__ xn) {
  __shared__ float red[256];
  const int row = blockIdx.x;
  const float* xr = x + row * EMB;
  float s = 0.f;
#pragma unroll
  for (int i = threadIdx.x; i < EMB; i += 256) { float v = xr[i]; s += v * v; }
  red[threadIdx.x] = s;
  __syncthreads();
  for (int st = 128; st > 0; st >>= 1) {
    if (threadIdx.x < st) red[threadIdx.x] += red[threadIdx.x + st];
    __syncthreads();
  }
  const float rinv = rsqrtf(red[0] * (1.f / EMB) + 1.1920929e-07f);
#pragma unroll
  for (int i = threadIdx.x; i < EMB; i += 256)
    xn[row * EMB + i] = (h16)(xr[i] * rinv * w[i]);
}

// ---------------------------------------------------------------------------
// C[M,N] = A[M,K](f16,row-major) * B[N,K](f16, Linear-weight layout) (+resid)
// Block tile 128(M) x 128(N), 8 waves in a 4(M) x 2(N) grid; each wave owns a
// 32x64 sub-tile = 8 f32 accumulators. LDS double-buffered; with async
// direct-to-LDS copies the k+1 tile streams in during the tile-k WMMAs.
__global__ void gemm_f16_kernel(const h16* __restrict__ A,
                                const h16* __restrict__ B,
                                const float* __restrict__ resid,
                                float* __restrict__ C, int N, int K) {
  __shared__ alignas(16) h16 sA[2][128 * 32];
  __shared__ alignas(16) h16 sB[2][128 * 32];
  const int tid = threadIdx.x;
  const int w   = tid >> 5;
  const int wm  = w & 3;   // M sub-tile: rows wm*32 .. wm*32+31
  const int wn  = w >> 2;  // N sub-tile: cols wn*64 .. wn*64+63
  const int gm0 = blockIdx.x * 128;
  const int gn0 = blockIdx.y * 128;
  const v8f z = {0.f, 0.f, 0.f, 0.f, 0.f, 0.f, 0.f, 0.f};
  v8f acc[2][4] = {{z, z, z, z}, {z, z, z, z}};

  auto stage = [&](int buf, int k0) {
#pragma unroll
    for (int t = 0; t < 2; ++t) {  // 128 rows x 4 chunks of 8 f16, each array
      int idx = tid + t * 256;
      int r = idx >> 2, c = idx & 3;
      cp16(&sA[buf][r * 32 + c * 8],
           A + (size_t)(gm0 + r) * K + k0 + c * 8);
      cp16(&sB[buf][r * 32 + c * 8],
           B + (size_t)(gn0 + r) * K + k0 + c * 8);
    }
  };

  stage(0, 0);
  cp_wait();
  __syncthreads();

  for (int k0 = 0; k0 < K; k0 += 32) {
    const int cur = (k0 >> 5) & 1;
    if (k0 + 32 < K) stage(cur ^ 1, k0 + 32);  // prefetch next tile

    v16h a[2];
#pragma unroll
    for (int im = 0; im < 2; ++im)
      a[im] = load_a_frag(&sA[cur][(wm * 32 + im * 16) * 32], 32);
#pragma unroll
    for (int jn = 0; jn < 4; ++jn) {
      v16h bf = load_b_frag(&sB[cur][(wn * 64 + jn * 16) * 32], 32);
#pragma unroll
      for (int im = 0; im < 2; ++im)
        acc[im][jn] = wmma_f16(a[im], bf, acc[im][jn]);
    }

    cp_wait();
    __syncthreads();
  }

  const int l = tid & 31, nn = l & 15, mh = (l >> 4) << 3;
#pragma unroll
  for (int im = 0; im < 2; ++im)
#pragma unroll
    for (int jn = 0; jn < 4; ++jn)
#pragma unroll
      for (int r = 0; r < 8; ++r) {
        size_t row = gm0 + wm * 32 + im * 16 + mh + r;
        size_t col = gn0 + wn * 64 + jn * 16 + nn;
        float v = acc[im][jn][r];
        if (resid) v += resid[row * N + col];
        C[row * N + col] = v;
      }
}

// ---------------------------------------------------------------------------
// Split QKV, apply rotation to q/k, write q/k as [b,h,s,d] f16 and v
// transposed [b,h,d,s] f16.
__global__ void rope_split_kernel(const float* __restrict__ qkv,
                                  const float* __restrict__ cosb,
                                  const float* __restrict__ sinb,
                                  h16* __restrict__ q, h16* __restrict__ k,
                                  h16* __restrict__ vt) {
  int t = blockIdx.x * 256 + threadIdx.x;  // over 2*2048*16*64
  int d  = t & 63;
  int hh = (t >> 6) & 15;
  int s  = (t >> 10) & 2047;
  int b  = t >> 21;
  const float* row = qkv + (size_t)(b * SEQ + s) * 3072;
  int d2 = (d + 32) & 63;
  float c  = cosb[s * 64 + d];
  float sn = sinb[s * 64 + d];
  float qv = row[hh * 64 + d],        qs = row[hh * 64 + d2];
  float kv = row[1024 + hh * 64 + d], ks = row[1024 + hh * 64 + d2];
  float vv = row[2048 + hh * 64 + d];
  int bh = b * NH + hh;
  q[(size_t)(bh * SEQ + s) * 64 + d]  = (h16)(c * qv + sn * qs);
  k[(size_t)(bh * SEQ + s) * 64 + d]  = (h16)(c * kv + sn * ks);
  vt[(size_t)(bh * 64 + d) * SEQ + s] = (h16)vv;
}

// ---------------------------------------------------------------------------
// Flash attention: block = (b, h, 128 q rows); 8 waves, each 16 q rows.
// Key/value blocks of 64 staged in LDS; online softmax per wave.
__global__ void attn_kernel(const h16* __restrict__ Q, const h16* __restrict__ Kc,
                            const h16* __restrict__ Vt, h16* __restrict__ O) {
  __shared__ alignas(16) h16 sK[64 * 64];      // [key][d]
  __shared__ alignas(16) h16 sV[64 * 64];      // [d][key]
  __shared__ alignas(16) h16 sP[8 * 16 * 64];  // per-wave P tiles [m][key]
  const int tid = threadIdx.x;
  const int w = tid >> 5, l = tid & 31;
  const int nn = l & 15, mh = (l >> 4) << 3;
  const int bx = blockIdx.x;
  const int qblk = bx & 15, h = (bx >> 4) & 15, b = bx >> 8;
  const int bh = b * NH + h;
  const int qt0 = qblk * 128 + w * 16;

  const h16* qbase = Q + (size_t)(bh * SEQ + qt0) * 64;
  const v16h aq0 = load_a_frag(qbase, 64);
  const v16h aq1 = load_a_frag(qbase + 32, 64);

  const v8f z = {0.f, 0.f, 0.f, 0.f, 0.f, 0.f, 0.f, 0.f};
  v8f o[4] = {z, z, z, z};
  float mrow[8], lrow[8];
#pragma unroll
  for (int r = 0; r < 8; ++r) { mrow[r] = -__builtin_inff(); lrow[r] = 0.f; }

  const int nkb = qblk * 2 + 2;
  for (int kb = 0; kb < nkb; ++kb) {
    const int kb0 = kb * 64;
#pragma unroll
    for (int t = 0; t < 2; ++t) {  // stage K (row-major) and Vt ([d][s])
      int idx = tid + t * 256;
      int r = idx >> 3, c = idx & 7;
      cp16(&sK[r * 64 + c * 8],
           Kc + (size_t)(bh * SEQ + kb0 + r) * 64 + c * 8);
      cp16(&sV[r * 64 + c * 8],
           Vt + (size_t)(bh * 64 + r) * SEQ + kb0 + c * 8);
    }
    cp_wait();
    __syncthreads();

    if (kb0 <= qt0 + 15) {
      v8f s4[4] = {z, z, z, z};
#pragma unroll
      for (int j = 0; j < 4; ++j) {
        v16h b0 = load_b_frag(&sK[(j * 16) * 64], 64);
        s4[j] = wmma_f16(aq0, b0, s4[j]);
        v16h b1 = load_b_frag(&sK[(j * 16) * 64 + 32], 64);
        s4[j] = wmma_f16(aq1, b1, s4[j]);
      }
      // scale + causal mask
#pragma unroll
      for (int j = 0; j < 4; ++j)
#pragma unroll
        for (int r = 0; r < 8; ++r) {
          float v = s4[j][r] * 0.125f;
          int row = qt0 + mh + r, col = kb0 + j * 16 + nn;
          s4[j][r] = (col > row) ? -__builtin_inff() : v;
        }
      // online softmax stats (rows live across 16-lane halves)
      float mnew[8], corr[8];
#pragma unroll
      for (int r = 0; r < 8; ++r) {
        float x = fmaxf(fmaxf(s4[0][r], s4[1][r]), fmaxf(s4[2][r], s4[3][r]));
#pragma unroll
        for (int off = 1; off <= 8; off <<= 1) x = fmaxf(x, __shfl_xor(x, off));
        mnew[r] = fmaxf(mrow[r], x);
        corr[r] = __expf(mrow[r] - mnew[r]);
      }
#pragma unroll
      for (int j = 0; j < 4; ++j)
#pragma unroll
        for (int r = 0; r < 8; ++r) s4[j][r] = __expf(s4[j][r] - mnew[r]);
#pragma unroll
      for (int r = 0; r < 8; ++r) {
        float sm = (s4[0][r] + s4[1][r]) + (s4[2][r] + s4[3][r]);
#pragma unroll
        for (int off = 1; off <= 8; off <<= 1) sm += __shfl_xor(sm, off);
        lrow[r] = lrow[r] * corr[r] + sm;
        mrow[r] = mnew[r];
      }
      // C-layout -> A-layout via per-wave LDS bounce (DS is in-order per wave)
      h16* pw = &sP[w * 1024];
#pragma unroll
      for (int j = 0; j < 4; ++j)
#pragma unroll
        for (int r = 0; r < 8; ++r)
          pw[(mh + r) * 64 + j * 16 + nn] = (h16)s4[j][r];
      asm volatile("s_wait_dscnt 0" ::: "memory");
#pragma unroll
      for (int jd = 0; jd < 4; ++jd)
#pragma unroll
        for (int r = 0; r < 8; ++r) o[jd][r] *= corr[r];
#pragma unroll
      for (int c = 0; c < 2; ++c) {
        v16h ap = load_a_frag(pw + c * 32, 64);
#pragma unroll
        for (int jd = 0; jd < 4; ++jd) {
          v16h bv = load_b_frag(&sV[(jd * 16) * 64 + c * 32], 64);
          o[jd] = wmma_f16(ap, bv, o[jd]);
        }
      }
    }
    __syncthreads();
  }

  // epilogue: O / l, write [b, s, h*64+d] f16 for the output projection
#pragma unroll
  for (int jd = 0; jd < 4; ++jd)
#pragma unroll
    for (int r = 0; r < 8; ++r) {
      size_t row = (size_t)b * SEQ + qt0 + mh + r;
      O[row * EMB + h * 64 + jd * 16 + nn] = (h16)(o[jd][r] / lrow[r]);
    }
}

// ---------------------------------------------------------------------------
extern "C" void kernel_launch(void* const* d_in, const int* in_sizes, int n_in,
                              void* d_out, int out_size, void* d_ws,
                              size_t ws_size, hipStream_t stream) {
  const float* emb  = (const float*)d_in[0];  // [2,2048,1024]
  const float* cosb = (const float*)d_in[1];  // [2048,64]
  const float* sinb = (const float*)d_in[2];
  const float* nw   = (const float*)d_in[3];  // [1024]
  const float* qkvw = (const float*)d_in[4];  // [3072,1024]
  const float* outw = (const float*)d_in[5];  // [1024,1024]
  float* out = (float*)d_out;

  const int ROWS = 2 * SEQ;  // 4096
  char* p = (char*)d_ws;
  auto take = [&](size_t bytes) {
    char* r = p;
    p += (bytes + 255) & ~(size_t)255;
    return r;
  };
  h16*   wq   = (h16*)take((size_t)3072 * 1024 * 2);
  h16*   wo   = (h16*)take((size_t)1024 * 1024 * 2);
  h16*   xn   = (h16*)take((size_t)ROWS * EMB * 2);
  float* qkvb = (float*)take((size_t)ROWS * 3072 * 4);
  h16*   qb   = (h16*)take((size_t)2 * NH * SEQ * 64 * 2);
  h16*   kb   = (h16*)take((size_t)2 * NH * SEQ * 64 * 2);
  h16*   vtb  = (h16*)take((size_t)2 * NH * 64 * SEQ * 2);
  h16*   attb = xn;  // xn dead after the QKV GEMM; reuse

  f32_to_f16_kernel<<<(3072 * 1024) / 256, 256, 0, stream>>>(qkvw, wq,
                                                             3072 * 1024);
  f32_to_f16_kernel<<<(1024 * 1024) / 256, 256, 0, stream>>>(outw, wo,
                                                             1024 * 1024);
  rmsnorm_kernel<<<ROWS, 256, 0, stream>>>(emb, nw, xn);
  gemm_f16_kernel<<<dim3(ROWS / 128, 3072 / 128), 256, 0, stream>>>(
      xn, wq, nullptr, qkvb, 3072, 1024);
  rope_split_kernel<<<(2 * SEQ * NH * 64) / 256, 256, 0, stream>>>(
      qkvb, cosb, sinb, qb, kb, vtb);
  attn_kernel<<<2 * NH * (SEQ / 128), 256, 0, stream>>>(qb, kb, vtb, attb);
  gemm_f16_kernel<<<dim3(ROWS / 128, EMB / 128), 256, 0, stream>>>(
      attb, wo, emb, out, EMB, 1024);
}
